// DFA_semseg_50165218018032
// MI455X (gfx1250) — compile-verified, matrix operand order checked by reading, not run on hardware
//
#include <hip/hip_runtime.h>
#include <hip/hip_bf16.h>

// ---------------------------------------------------------------------------
// DGCNN-semseg forward for MI455X (gfx1250, wave32).
//  * 1x1-conv layers -> v_wmma_f32_16x16x32_bf16 (bf16 in, f32 accum),
//    32(o) x 32(s) tile per wave = 4 WMMA per K-step, A/B fragments reused.
//  * All hot-loop addressing in 32-bit element offsets (tensors < 2^31 elems).
//  * KNN in f32 VALU (top-k selection is precision sensitive).
//  * BN training-mode: GEMM -> channel stats -> fused scale/shift+lrelu.
// ---------------------------------------------------------------------------

typedef __attribute__((ext_vector_type(16))) __bf16 v16bf;
typedef __attribute__((ext_vector_type(8)))  float  v8f;

#define BATCH 2
#define NPTS  4096
#define KNN_K 20
#define NKF   (NPTS * KNN_K)
#define NEG_BIG (-1.0e30f)

// ---------------------------------------------------------------------------
// WMMA GEMM:  Y[b, yc0+o, s] = sum_c W[o, c] * X[b, xc0+c, s]
// One wave per 32x32 output tile. Requires S % 32 == 0 (true at all sites).
// ---------------------------------------------------------------------------
__global__ void wmma_gemm(const float* __restrict__ W, int Cout, int Cin,
                          const float* __restrict__ X, int Xs, int xc0,
                          float* __restrict__ Y, int Ys, int yc0, int S)
{
    const int sBase = blockIdx.x * 32;
    const int oBase = blockIdx.y * 32;
    const int bb    = blockIdx.z;
    const int lane  = threadIdx.x & 31;
    const int mn    = lane & 15;
    const int half  = lane >> 4;

    // clamped weight rows: padded o-rows make garbage D rows we never store
    const int o0 = oBase + mn,      o1 = oBase + 16 + mn;
    const int r0 = (o0 < Cout) ? o0 : (Cout - 1);
    const int r1 = (o1 < Cout) ? o1 : (Cout - 1);
    const float* __restrict__ Wr0 = W + r0 * Cin;
    const float* __restrict__ Wr1 = W + r1 * Cin;
    const int xbase = (bb * Xs + xc0) * S + sBase + mn;   // 32-bit offsets

    v8f acc00 = {}, acc01 = {}, acc10 = {}, acc11 = {};

    const int cfull = Cin & ~31;
    // ---- unguarded steady-state K loop ----
    for (int c0 = 0; c0 < cfull; c0 += 32) {
        v16bf a0, a1, b0, b1;
#pragma unroll
        for (int j = 0; j < 16; ++j) {
            const int kk = ((j >> 3) << 4) + (half << 3) + (j & 7);
            const int c  = c0 + kk;
            const int xo = xbase + c * S;
            a0[j] = (__bf16)Wr0[c];
            a1[j] = (__bf16)Wr1[c];
            b0[j] = (__bf16)X[xo];
            b1[j] = (__bf16)X[xo + 16];
        }
        acc00 = __builtin_amdgcn_wmma_f32_16x16x32_bf16(
                    false, a0, false, b0, (short)0, acc00, false, false);
        acc01 = __builtin_amdgcn_wmma_f32_16x16x32_bf16(
                    false, a0, false, b1, (short)0, acc01, false, false);
        acc10 = __builtin_amdgcn_wmma_f32_16x16x32_bf16(
                    false, a1, false, b0, (short)0, acc10, false, false);
        acc11 = __builtin_amdgcn_wmma_f32_16x16x32_bf16(
                    false, a1, false, b1, (short)0, acc11, false, false);
    }
    // ---- masked tail (only layers with Cin % 32 != 0); zero B side only ----
    if (cfull < Cin) {
        v16bf a0, a1, b0, b1;
#pragma unroll
        for (int j = 0; j < 16; ++j) {
            const int kk = ((j >> 3) << 4) + (half << 3) + (j & 7);
            const int c  = cfull + kk;
            const int cc = (c < Cin) ? c : (Cin - 1);
            const float bm = (c < Cin) ? 1.0f : 0.0f;
            const int xo = xbase + cc * S;
            a0[j] = (__bf16)Wr0[cc];
            a1[j] = (__bf16)Wr1[cc];
            b0[j] = (__bf16)(X[xo] * bm);
            b1[j] = (__bf16)(X[xo + 16] * bm);
        }
        acc00 = __builtin_amdgcn_wmma_f32_16x16x32_bf16(
                    false, a0, false, b0, (short)0, acc00, false, false);
        acc01 = __builtin_amdgcn_wmma_f32_16x16x32_bf16(
                    false, a0, false, b1, (short)0, acc01, false, false);
        acc10 = __builtin_amdgcn_wmma_f32_16x16x32_bf16(
                    false, a1, false, b0, (short)0, acc10, false, false);
        acc11 = __builtin_amdgcn_wmma_f32_16x16x32_bf16(
                    false, a1, false, b1, (short)0, acc11, false, false);
    }

    const int ybase = (bb * Ys + yc0 + oBase) * S + sBase + mn;
    if (oBase + 32 <= Cout) {           // fast path: whole 32-row block valid
#pragma unroll
        for (int r = 0; r < 8; ++r) {
            const int row = (r + (half << 3)) * S;
            Y[ybase + row]            = acc00[r];
            Y[ybase + row + 16]       = acc01[r];
            Y[ybase + row + 16 * S]      = acc10[r];
            Y[ybase + row + 16 * S + 16] = acc11[r];
        }
    } else {
#pragma unroll
        for (int r = 0; r < 8; ++r) {
            const int m   = r + (half << 3);
            const int row = m * S;
            if (oBase + m < Cout) {
                Y[ybase + row]      = acc00[r];
                Y[ybase + row + 16] = acc01[r];
            }
            if (oBase + 16 + m < Cout) {
                Y[ybase + row + 16 * S]      = acc10[r];
                Y[ybase + row + 16 * S + 16] = acc11[r];
            }
        }
    }
}

// ---------------------------------------------------------------------------
// BN stats: per channel mean/var over (batch, spatial) -> scale/shift.
// ---------------------------------------------------------------------------
__global__ void bn_stats(const float* __restrict__ Y, int Ys, int yc0,
                         int S, const float* __restrict__ g,
                         const float* __restrict__ bt,
                         float* __restrict__ scale, float* __restrict__ shift)
{
    const int o = blockIdx.x;
    const int t = threadIdx.x;
    __shared__ float s1s[256], s2s[256];
    float s1 = 0.f, s2 = 0.f;
    const int tot = BATCH * S;
    for (int i = t; i < tot; i += 256) {
        const int bb = i / S, s = i % S;
        const float v = Y[(bb * Ys + yc0 + o) * S + s];
        s1 += v; s2 += v * v;
    }
    s1s[t] = s1; s2s[t] = s2; __syncthreads();
    for (int off = 128; off > 0; off >>= 1) {
        if (t < off) { s1s[t] += s1s[t + off]; s2s[t] += s2s[t + off]; }
        __syncthreads();
    }
    if (t == 0) {
        const float inv_n = 1.0f / (float)tot;
        const float mean  = s1s[0] * inv_n;
        const float var   = s2s[0] * inv_n - mean * mean;
        const float inv   = rsqrtf(var + 1e-5f);
        const float sc    = g[o] * inv;
        scale[o] = sc;
        shift[o] = bt[o] - mean * sc;
    }
}

__global__ void bn_lrelu(float* __restrict__ Y, int Ys, int yc0, int C, int S,
                         const float* __restrict__ scale,
                         const float* __restrict__ shift)
{
    const size_t i = (size_t)blockIdx.x * 256 + threadIdx.x;
    const size_t tot = (size_t)BATCH * C * S;
    if (i >= tot) return;
    const int s  = (int)(i % S);
    const int c  = (int)((i / S) % C);
    const int bb = (int)(i / ((size_t)S * C));
    const int p  = (bb * Ys + yc0 + c) * S + s;
    const float v = Y[p] * scale[c] + shift[c];
    Y[p] = (v < 0.f) ? 0.2f * v : v;
}

// ---------------------------------------------------------------------------
// Fused KNN top-20 on pd = 2 q_n.q_m - |q_n|^2 - |q_m|^2 (larger = closer).
// One workgroup per (point, batch); per-thread sorted top-20, LDS head-merge.
// ---------------------------------------------------------------------------
__global__ void knn_topk(const float* __restrict__ X, int Xs, int coff, int C,
                         float* __restrict__ dis, int* __restrict__ idx)
{
    const int n = blockIdx.x, b = blockIdx.y, t = threadIdx.x;
    __shared__ float qn[64];
    __shared__ float lv[256 * KNN_K];
    __shared__ int   li[256 * KNN_K];
    __shared__ float rv[256];
    __shared__ int   ri[256];
    __shared__ int   headS[256];

    if (t < C) qn[t] = X[(b * Xs + coff + t) * NPTS + n];
    __syncthreads();

    float xxn = 0.f;
    for (int c = 0; c < C; ++c) xxn += qn[c] * qn[c];

    float val[KNN_K]; int vid[KNN_K];
#pragma unroll
    for (int i = 0; i < KNN_K; ++i) { val[i] = NEG_BIG; vid[i] = 0; }

    for (int m = t; m < NPTS; m += 256) {
        float dot = 0.f, xxm = 0.f;
        for (int c = 0; c < C; ++c) {
            const float v = X[(b * Xs + coff + c) * NPTS + m];
            dot += qn[c] * v; xxm += v * v;
        }
        const float d = 2.0f * dot - xxn - xxm;
        if (d > val[KNN_K - 1]) {
            int p = KNN_K - 1;
            while (p > 0 && val[p - 1] < d) {
                val[p] = val[p - 1]; vid[p] = vid[p - 1]; --p;
            }
            val[p] = d; vid[p] = m;
        }
    }
#pragma unroll
    for (int i = 0; i < KNN_K; ++i) { lv[t * KNN_K + i] = val[i]; li[t * KNN_K + i] = vid[i]; }
    headS[t] = 0;
    __syncthreads();

    for (int r = 0; r < KNN_K; ++r) {
        const int h = headS[t];
        rv[t] = (h < KNN_K) ? lv[t * KNN_K + h] : NEG_BIG;
        ri[t] = t;
        __syncthreads();
        for (int off = 128; off > 0; off >>= 1) {
            if (t < off && rv[t + off] > rv[t]) { rv[t] = rv[t + off]; ri[t] = ri[t + off]; }
            __syncthreads();
        }
        if (t == 0) {
            const int w  = ri[0];
            const int h2 = headS[w];
            dis[(b * NPTS + n) * KNN_K + r] = lv[w * KNN_K + h2];
            idx[(b * NPTS + n) * KNN_K + r] = li[w * KNN_K + h2];
            headS[w] = h2 + 1;
        }
        __syncthreads();
    }
}

// ---------------------------------------------------------------------------
// Graph feature: G[b, gc0+ch, n, kk] = (ch<C) ? X[nbr]-X[ctr] : X[ctr]
// ---------------------------------------------------------------------------
__global__ void build_feat(const float* __restrict__ X, int Xs, int xc0, int C,
                           const int* __restrict__ idx,
                           float* __restrict__ G, int Gs, int gc0)
{
    const size_t i = (size_t)blockIdx.x * 256 + threadIdx.x;
    const size_t tot = (size_t)BATCH * 2 * C * NKF;
    if (i >= tot) return;
    const int kk = (int)(i % KNN_K);
    const int n  = (int)((i / KNN_K) % NPTS);
    const int ch = (int)((i / NKF) % (2 * C));
    const int b  = (int)(i / ((size_t)NKF * 2 * C));
    float v;
    if (ch < C) {
        const int m = idx[(b * NPTS + n) * KNN_K + kk];
        v = X[(b * Xs + xc0 + ch) * NPTS + m]
          - X[(b * Xs + xc0 + ch) * NPTS + n];
    } else {
        v = X[(b * Xs + xc0 + ch - C) * NPTS + n];
    }
    G[(b * Gs + gc0 + ch) * NKF + n * KNN_K + kk] = v;
}

// Geometric tensor: [ctr(3), nbr(3), ctr-nbr(3), dis(1)] -> 10 channels
__global__ void build_geo(const float* __restrict__ xyz, int Xs,
                          const int* __restrict__ idx,
                          const float* __restrict__ dis,
                          float* __restrict__ Cb, int Cs)
{
    const size_t i = (size_t)blockIdx.x * 256 + threadIdx.x;
    const size_t tot = (size_t)BATCH * NKF;
    if (i >= tot) return;
    const int kk = (int)(i % KNN_K);
    const int n  = (int)((i / KNN_K) % NPTS);
    const int b  = (int)(i / NKF);
    const int m  = idx[(b * NPTS + n) * KNN_K + kk];
    const int sp = n * KNN_K + kk;
#pragma unroll
    for (int d = 0; d < 3; ++d) {
        const float ctr = xyz[(b * Xs + d) * NPTS + n];
        const float nbr = xyz[(b * Xs + d) * NPTS + m];
        Cb[(b * Cs + d)     * NKF + sp] = ctr;
        Cb[(b * Cs + 3 + d) * NKF + sp] = nbr;
        Cb[(b * Cs + 6 + d) * NKF + sp] = ctr - nbr;
    }
    Cb[(b * Cs + 9) * NKF + sp] = dis[(b * NPTS + n) * KNN_K + kk];
}

// max over neighbor axis k
__global__ void max_over_k(const float* __restrict__ Y, int Ys, int yc0, int C,
                           float* __restrict__ Out, int Os, int oc0)
{
    const size_t i = (size_t)blockIdx.x * 256 + threadIdx.x;
    const size_t tot = (size_t)BATCH * C * NPTS;
    if (i >= tot) return;
    const int n = (int)(i % NPTS);
    const int c = (int)((i / NPTS) % C);
    const int b = (int)(i / ((size_t)NPTS * C));
    const float* p = &Y[(b * Ys + yc0 + c) * NKF + n * KNN_K];
    float m = p[0];
#pragma unroll
    for (int kk = 1; kk < KNN_K; ++kk) m = fmaxf(m, p[kk]);
    Out[(b * Os + oc0 + c) * NPTS + n] = m;
}

// max over spatial axis N -> vals[b*C+c]
__global__ void max_over_n(const float* __restrict__ Y, int Ys, int yc0, int C,
                           float* __restrict__ vals)
{
    const int bc = blockIdx.x;       // b*C + c
    const int b = bc / C, c = bc % C, t = threadIdx.x;
    __shared__ float red[256];
    float m = NEG_BIG;
    for (int n = t; n < NPTS; n += 256)
        m = fmaxf(m, Y[(b * Ys + yc0 + c) * NPTS + n]);
    red[t] = m; __syncthreads();
    for (int off = 128; off > 0; off >>= 1) {
        if (t < off) red[t] = fmaxf(red[t], red[t + off]);
        __syncthreads();
    }
    if (t == 0) vals[bc] = red[0];
}

__global__ void bcast_fill(const float* __restrict__ vals, int C,
                           float* __restrict__ Out, int Os, int oc0)
{
    const size_t i = (size_t)blockIdx.x * 256 + threadIdx.x;
    const size_t tot = (size_t)BATCH * C * NPTS;
    if (i >= tot) return;
    const int n = (int)(i % NPTS);
    const int c = (int)((i / NPTS) % C);
    const int b = (int)(i / ((size_t)NPTS * C));
    Out[(b * Os + oc0 + c) * NPTS + n] = vals[b * C + c];
}

__global__ void copy_channels(const float* __restrict__ Src, int Ss, int sc0,
                              float* __restrict__ Dst, int Ds, int dc0, int C)
{
    const size_t i = (size_t)blockIdx.x * 256 + threadIdx.x;
    const size_t tot = (size_t)BATCH * C * NPTS;
    if (i >= tot) return;
    const int n = (int)(i % NPTS);
    const int c = (int)((i / NPTS) % C);
    const int b = (int)(i / ((size_t)NPTS * C));
    Dst[(b * Ds + dc0 + c) * NPTS + n] = Src[(b * Ss + sc0 + c) * NPTS + n];
}

// small dense layers on (B, C) vectors
__global__ void linear_small(const float* __restrict__ W,
                             const float* __restrict__ Xv,
                             float* __restrict__ Yv, int Cout, int Cin)
{
    const int i = blockIdx.x * 64 + threadIdx.x;
    if (i >= BATCH * Cout) return;
    const int b = i / Cout, o = i % Cout;
    float acc = 0.f;
    for (int c = 0; c < Cin; ++c)
        acc += W[o * Cin + c] * Xv[b * Cin + c];
    Yv[i] = acc;
}

__global__ void tfm_final(const float* __restrict__ W,
                          const float* __restrict__ bias,
                          const float* __restrict__ Xv, float* __restrict__ T)
{
    const int i = blockIdx.x * 32 + threadIdx.x;
    if (i >= BATCH * 9) return;
    const int b = i / 9, o = i % 9;
    float acc = bias[o];
    for (int c = 0; c < 256; ++c)
        acc += W[o * 256 + c] * Xv[b * 256 + c];
    T[i] = acc;
}

// xnew[b,d,n] = sum_c x[b,c,n]*T[b,c*3+d] (d<3), else passthrough
__global__ void apply_transform(const float* __restrict__ x,
                                const float* __restrict__ T,
                                float* __restrict__ xn)
{
    const size_t i = (size_t)blockIdx.x * 256 + threadIdx.x;
    const size_t tot = (size_t)BATCH * 9 * NPTS;
    if (i >= tot) return;
    const int n  = (int)(i % NPTS);
    const int ch = (int)((i / NPTS) % 9);
    const int b  = (int)(i / ((size_t)NPTS * 9));
    float v;
    if (ch < 3) {
        v = 0.f;
#pragma unroll
        for (int c = 0; c < 3; ++c)
            v += x[(b * 9 + c) * NPTS + n] * T[b * 9 + c * 3 + ch];
    } else {
        v = x[(b * 9 + ch) * NPTS + n];
    }
    xn[(b * 9 + ch) * NPTS + n] = v;
}

// ---------------------------------------------------------------------------
extern "C" void kernel_launch(void* const* d_in, const int* in_sizes, int n_in,
                              void* d_out, int out_size, void* d_ws, size_t ws_size,
                              hipStream_t stream)
{
    (void)in_sizes; (void)n_in; (void)out_size; (void)ws_size;
    const float* x = (const float*)d_in[0];
    auto Wp = [&](int l) { return (const float*)d_in[1 + 3 * l]; };
    auto Gp = [&](int l) { return (const float*)d_in[2 + 3 * l]; };
    auto Bp = [&](int l) { return (const float*)d_in[3 + 3 * l]; };
    const float* conv10_w = (const float*)d_in[61];
    const float* ttfm_w   = (const float*)d_in[62];
    const float* ttfm_b   = (const float*)d_in[63];
    float* out = (float*)d_out;

    // layer indices in params dict order
    enum { T1=0,T2,T3,TL1,TL2,P1,P2,M1,M2,M3,C1,C2,C3,C4,C5,CC,C6,C7,C8,C9 };

    // -------- workspace arena --------
    char* wsp = (char*)d_ws; size_t off = 0;
    auto allocf = [&](size_t n)->float* {
        float* p = (float*)(wsp + off); off += ((n * 4 + 255) / 256) * 256; return p; };
    auto alloci = [&](size_t n)->int* {
        int* p = (int*)(wsp + off); off += ((n * 4 + 255) / 256) * 256; return p; };

    float* disA = allocf((size_t)BATCH * NPTS * KNN_K);
    int*   idxA = alloci((size_t)BATCH * NPTS * KNN_K);
    float* disB = allocf((size_t)BATCH * NPTS * KNN_K);
    int*   idxB = alloci((size_t)BATCH * NPTS * KNN_K);
    float* disC = allocf((size_t)BATCH * NPTS * KNN_K);
    int*   idxC = alloci((size_t)BATCH * NPTS * KNN_K);
    float* feat18  = allocf((size_t)BATCH * 18  * NKF);
    float* big1    = allocf((size_t)BATCH * 256 * NKF);  // graph concat buffer
    float* big2    = allocf((size_t)BATCH * 128 * NKF);  // 128-ch-stride conv output (T2, C2, C3, C5)
    float* big3    = allocf((size_t)BATCH * 64  * NKF);  // 64-ch conv output (T1, C1, C4, CC)
    float* cbuf    = allocf((size_t)BATCH * 10  * NKF);
    float* t2m     = allocf((size_t)BATCH * 128 * NPTS);
    float* t3      = allocf((size_t)BATCH * 1024 * NPTS);
    float* vec1024 = allocf((size_t)BATCH * 1024);
    float* lin1v   = allocf((size_t)BATCH * 512);
    float* lin2v   = allocf((size_t)BATCH * 256);
    float* tmat    = allocf((size_t)BATCH * 16);
    float* xnew    = allocf((size_t)BATCH * 9  * NPTS);
    float* p1      = allocf((size_t)BATCH * 32 * NPTS);
    float* p2      = allocf((size_t)BATCH * 64 * NPTS);
    float* vec64   = allocf((size_t)BATCH * 64);
    float* cat256  = allocf((size_t)BATCH * 256 * NPTS);
    float* y6      = allocf((size_t)BATCH * 1024 * NPTS);
    float* vec1024b= allocf((size_t)BATCH * 1024);
    float* cat1216 = allocf((size_t)BATCH * 1216 * NPTS);
    float* y7      = allocf((size_t)BATCH * 512 * NPTS);
    float* y8      = allocf((size_t)BATCH * 256 * NPTS);
    float* y9      = allocf((size_t)BATCH * 128 * NPTS);
    float* sc      = allocf(1024);
    float* sh      = allocf(1024);

    auto gemm = [&](const float* W, int Cout, int Cin,
                    const float* X, int Xs, int xc0,
                    float* Y, int Ys, int yc0, int S) {
        dim3 g((unsigned)(S / 32), (unsigned)((Cout + 31) / 32), BATCH);
        wmma_gemm<<<g, 32, 0, stream>>>(W, Cout, Cin, X, Xs, xc0, Y, Ys, yc0, S);
    };
    auto bn = [&](int layer, float* Y, int Ys, int yc0, int C, int S) {
        bn_stats<<<C, 256, 0, stream>>>(Y, Ys, yc0, S, Gp(layer), Bp(layer), sc, sh);
        size_t tot = (size_t)BATCH * C * S;
        bn_lrelu<<<(unsigned)((tot + 255) / 256), 256, 0, stream>>>(Y, Ys, yc0, C, S, sc, sh);
    };
    auto blocks = [&](size_t tot) { return (unsigned)((tot + 255) / 256); };

    // ===== KNN #1 on x[:,6:9] (valid pre- and post-transform) =====
    knn_topk<<<dim3(NPTS, BATCH), 256, 0, stream>>>(x, 9, 6, 3, disA, idxA);

    // ===== transform net =====
    build_feat<<<blocks((size_t)BATCH * 18 * NKF), 256, 0, stream>>>(
        x, 9, 0, 9, idxA, feat18, 18, 0);
    gemm(Wp(T1), 64, 18, feat18, 18, 0, big3, 64, 0, NKF);   bn(T1, big3, 64, 0, 64, NKF);
    gemm(Wp(T2), 128, 64, big3, 64, 0, big2, 128, 0, NKF);   bn(T2, big2, 128, 0, 128, NKF);
    max_over_k<<<blocks((size_t)BATCH * 128 * NPTS), 256, 0, stream>>>(
        big2, 128, 0, 128, t2m, 128, 0);
    gemm(Wp(T3), 1024, 128, t2m, 128, 0, t3, 1024, 0, NPTS); bn(T3, t3, 1024, 0, 1024, NPTS);
    max_over_n<<<BATCH * 1024, 256, 0, stream>>>(t3, 1024, 0, 1024, vec1024);
    linear_small<<<(BATCH * 512 + 63) / 64, 64, 0, stream>>>(Wp(TL1), vec1024, lin1v, 512, 1024);
    bn(TL1, lin1v, 512, 0, 512, 1);
    linear_small<<<(BATCH * 256 + 63) / 64, 64, 0, stream>>>(Wp(TL2), lin1v, lin2v, 256, 512);
    bn(TL2, lin2v, 256, 0, 256, 1);
    tfm_final<<<1, 32, 0, stream>>>(ttfm_w, ttfm_b, lin2v, tmat);
    apply_transform<<<blocks((size_t)BATCH * 9 * NPTS), 256, 0, stream>>>(x, tmat, xnew);

    // ===== point MLP: x0 =====
    gemm(Wp(P1), 32, 9, xnew, 9, 0, p1, 32, 0, NPTS);  bn(P1, p1, 32, 0, 32, NPTS);
    gemm(Wp(P2), 64, 32, p1, 32, 0, p2, 64, 0, NPTS);  bn(P2, p2, 64, 0, 64, NPTS);
    max_over_n<<<BATCH * 64, 256, 0, stream>>>(p2, 64, 0, 64, vec64);
    bcast_fill<<<blocks((size_t)BATCH * 64 * NPTS), 256, 0, stream>>>(vec64, 64, cat256, 256, 0);

    // ===== stage 1 (reuses idxA/disA) =====
    build_geo<<<blocks((size_t)BATCH * NKF), 256, 0, stream>>>(x, 9, idxA, disA, cbuf, 10);
    gemm(Wp(M1), 18, 10, cbuf, 10, 0, big1, 36, 0, NKF);  bn(M1, big1, 36, 0, 18, NKF);
    build_feat<<<blocks((size_t)BATCH * 18 * NKF), 256, 0, stream>>>(
        xnew, 9, 0, 9, idxA, big1, 36, 18);
    gemm(Wp(C1), 64, 36, big1, 36, 0, big3, 64, 0, NKF);  bn(C1, big3, 64, 0, 64, NKF);
    gemm(Wp(C2), 64, 64, big3, 64, 0, big2, 128, 0, NKF); bn(C2, big2, 128, 0, 64, NKF);
    max_over_k<<<blocks((size_t)BATCH * 64 * NPTS), 256, 0, stream>>>(
        big2, 128, 0, 64, cat256, 256, 64);                // x1 -> cat256[64:128]

    // ===== stage 2 (knn on x1) =====
    knn_topk<<<dim3(NPTS, BATCH), 256, 0, stream>>>(cat256, 256, 64, 64, disB, idxB);
    build_geo<<<blocks((size_t)BATCH * NKF), 256, 0, stream>>>(x, 9, idxB, disB, cbuf, 10);
    gemm(Wp(M2), 128, 10, cbuf, 10, 0, big1, 256, 0, NKF); bn(M2, big1, 256, 0, 128, NKF);
    build_feat<<<blocks((size_t)BATCH * 128 * NKF), 256, 0, stream>>>(
        cat256, 256, 64, 64, idxB, big1, 256, 128);
    gemm(Wp(C3), 128, 256, big1, 256, 0, big2, 128, 0, NKF); bn(C3, big2, 128, 0, 128, NKF);
    gemm(Wp(C4), 64, 128, big2, 128, 0, big3, 64, 0, NKF);   bn(C4, big3, 64, 0, 64, NKF);
    max_over_k<<<blocks((size_t)BATCH * 64 * NPTS), 256, 0, stream>>>(
        big3, 64, 0, 64, cat256, 256, 128);                // x2 -> cat256[128:192]

    // ===== stage 3 (knn on x2) =====
    knn_topk<<<dim3(NPTS, BATCH), 256, 0, stream>>>(cat256, 256, 128, 64, disC, idxC);
    build_geo<<<blocks((size_t)BATCH * NKF), 256, 0, stream>>>(x, 9, idxC, disC, cbuf, 10);
    gemm(Wp(M3), 128, 10, cbuf, 10, 0, big1, 256, 0, NKF); bn(M3, big1, 256, 0, 128, NKF);
    build_feat<<<blocks((size_t)BATCH * 128 * NKF), 256, 0, stream>>>(
        cat256, 256, 128, 64, idxC, big1, 256, 128);
    gemm(Wp(C5), 128, 256, big1, 256, 0, big2, 128, 0, NKF); bn(C5, big2, 128, 0, 128, NKF);
    gemm(Wp(CC), 64, 128, big2, 128, 0, big3, 64, 0, NKF);   bn(CC, big3, 64, 0, 64, NKF);
    max_over_k<<<blocks((size_t)BATCH * 64 * NPTS), 256, 0, stream>>>(
        big3, 64, 0, 64, cat256, 256, 192);                // x3 -> cat256[192:256]

    // ===== global aggregation =====
    gemm(Wp(C6), 1024, 256, cat256, 256, 0, y6, 1024, 0, NPTS); bn(C6, y6, 1024, 0, 1024, NPTS);
    max_over_n<<<BATCH * 1024, 256, 0, stream>>>(y6, 1024, 0, 1024, vec1024b);
    bcast_fill<<<blocks((size_t)BATCH * 1024 * NPTS), 256, 0, stream>>>(
        vec1024b, 1024, cat1216, 1216, 0);
    copy_channels<<<blocks((size_t)BATCH * 192 * NPTS), 256, 0, stream>>>(
        cat256, 256, 64, cat1216, 1216, 1024, 192);        // x1|x2|x3

    gemm(Wp(C7), 512, 1216, cat1216, 1216, 0, y7, 512, 0, NPTS); bn(C7, y7, 512, 0, 512, NPTS);
    gemm(Wp(C8), 256, 512, y7, 512, 0, y8, 256, 0, NPTS);        bn(C8, y8, 256, 0, 256, NPTS);
    gemm(Wp(C9), 128, 256, y8, 256, 0, y9, 128, 0, NPTS);        bn(C9, y9, 128, 0, 128, NPTS);
    gemm(conv10_w, 13, 128, y9, 128, 0, out, 13, 0, NPTS);       // no BN on output
}
